// SSAML_75239237092001
// MI455X (gfx1250) — matrix operand order repaired and checked by reading, not compiled, not run on hardware
//
#include <hip/hip_runtime.h>

typedef __bf16 v16bf __attribute__((ext_vector_type(16)));
typedef float  v8f   __attribute__((ext_vector_type(8)));
typedef int    gvec4i __attribute__((vector_size(16)));   // matches async-LDS builtin pointee

union FragB16 {
    v16bf v;
    __bf16 h[16];
    unsigned int u[8];
};

constexpr int Bn = 8, Cc = 256, Nn = 2304;
constexpr size_t QELEMS = (size_t)Bn * Nn * Cc;   // elems per Q/K/V
constexpr int XS_STRIDE = 264;                    // padded bf16 row stride (x tile)
constexpr int KS_STRIDE = 264;                    // padded bf16 row stride (K tile)
constexpr int VT_STRIDE = 40;                     // padded bf16 row stride (V^T tile)
constexpr int OF_STRIDE = 257;                    // padded f32 row stride (out tile)

// Dynamic LDS layout for the attention kernel (bytes)
constexpr int KSB_BYTES   = 32 * KS_STRIDE * 2;           // 16,896
constexpr int VRAW_BYTES  = 32 * Cc * 2;                  // 16,384
constexpr int OFF_KS0     = 0;                            // two K buffers follow each other
constexpr int OFF_VR0     = OFF_KS0 + 2 * KSB_BYTES;      // 33,792 (two Vraw buffers)
constexpr int OFF_VT      = OFF_VR0 + 2 * VRAW_BYTES;     // 66,560
constexpr int OFF_PS      = OFF_VT + 256 * VT_STRIDE * 2; // 87,040
constexpr int ATTN_LDS    = OFF_PS + 4 * 16 * 32 * 2;     // 91,136
constexpr int EPI_LDS     = 64 * OF_STRIDE * 4;           // 65,792 (aliases the above)
constexpr int ATTN_LDS_TOTAL = (ATTN_LDS > EPI_LDS) ? ATTN_LDS : EPI_LDS;

// A-matrix (16x32 bf16) per-lane K index for VGPR v, lane-group g (ISA 7.12.2)
__device__ __forceinline__ int a_k(int v, int g) {
    return (v < 4) ? (2 * v + 8 * g) : (16 + 2 * (v - 4) + 8 * g);
}

// ---- CDNA5 async global->LDS copy (ASYNCcnt-tracked), with sync fallback ----
#if __has_builtin(__builtin_amdgcn_global_load_async_to_lds_b128)
#define HAS_ASYNC_LDS 1
#else
#define HAS_ASYNC_LDS 0
#endif

__device__ __forceinline__ void async_copy16(const void* g, void* l) {
#if HAS_ASYNC_LDS
    __builtin_amdgcn_global_load_async_to_lds_b128(
        (__attribute__((address_space(1))) gvec4i*)g,
        (__attribute__((address_space(3))) gvec4i*)l, 0, 0);
#else
    *(uint4*)l = *(const uint4*)g;
#endif
}

#if HAS_ASYNC_LDS
#if __has_builtin(__builtin_amdgcn_s_wait_asynccnt)
#define WAIT_ASYNC(n) __builtin_amdgcn_s_wait_asynccnt(n)
#else
#define WAIT_ASYNC(n) asm volatile("s_wait_asynccnt %0" ::"i"(n) : "memory")
#endif
#else
#define WAIT_ASYNC(n)
#endif

// ---------------------------------------------------------------------------
// Kernel 1: weights/bias -> bf16 workspace. Wq and bq are NEGATED so that the
// attention kernel computes S = -(q.k) directly (softmax(max-E) == softmax(-E)).
// ---------------------------------------------------------------------------
__global__ void SSAML_prep_kernel(const float* __restrict__ Wq, const float* __restrict__ bq,
                                  const float* __restrict__ Wk, const float* __restrict__ bk,
                                  const float* __restrict__ Wv, const float* __restrict__ bv,
                                  unsigned short* __restrict__ wbf_, float* __restrict__ bias) {
    __bf16* wbf = reinterpret_cast<__bf16*>(wbf_);
    int idx = blockIdx.x * 256 + threadIdx.x;
    if (idx < 3 * Cc * Cc) {
        int mat = idx / (Cc * Cc);
        int r   = idx - mat * Cc * Cc;
        const float* W = (mat == 0) ? Wq : ((mat == 1) ? Wk : Wv);
        float val = W[r];
        if (mat == 0) val = -val;
        wbf[idx] = (__bf16)val;
    }
    if (idx < 3 * Cc) {
        int mat = idx / Cc;
        int j   = idx - mat * Cc;
        const float* bb = (mat == 0) ? bq : ((mat == 1) ? bk : bv);
        float val = bb[j];
        if (mat == 0) val = -val;
        bias[idx] = val;
    }
}

// ---------------------------------------------------------------------------
// Kernel 2: QKV projection.  xf(n,c) = x(b,c,n);  out[mat] = xf @ W^T + b.
// Block = 128 threads (4 waves), 64 tokens per block, 16 tokens per wave.
// ---------------------------------------------------------------------------
__global__ __launch_bounds__(128) void SSAML_qkv_kernel(const float* __restrict__ x,
        const unsigned short* __restrict__ wbf_, const float* __restrict__ bias,
        unsigned short* __restrict__ qkv_) {
    const __bf16* wbf = reinterpret_cast<const __bf16*>(wbf_);
    __bf16* qkv = reinterpret_cast<__bf16*>(qkv_);
    __shared__ __align__(16) __bf16 Xs[64 * XS_STRIDE];

    const int b  = blockIdx.x / (Nn / 64);
    const int n0 = (blockIdx.x % (Nn / 64)) * 64;
    const int t  = threadIdx.x;
    const float* xb = x + (size_t)b * Cc * Nn;

    for (int j = 0; j < 128; ++j) {
        int idx = j * 128 + t;
        int c = idx >> 6, n = idx & 63;
        Xs[n * XS_STRIDE + c] = (__bf16)xb[(size_t)c * Nn + n0 + n];
    }
    __syncthreads();

    const int lane = t & 31, wave = t >> 5;
    const int g = lane >> 4, ln = lane & 15;
    const int row0 = wave * 16;

    FragB16 af[8];
#pragma unroll
    for (int s = 0; s < 8; ++s)
#pragma unroll
        for (int v = 0; v < 8; ++v)
            af[s].u[v] = *(const unsigned int*)&Xs[(row0 + ln) * XS_STRIDE + s * 32 + a_k(v, g)];

    for (int mat = 0; mat < 3; ++mat) {
        const __bf16* wm = wbf + (size_t)mat * Cc * Cc;
        __bf16* om = qkv + (size_t)mat * QELEMS + ((size_t)b * Nn + n0 + row0) * Cc;
        for (int jt = 0; jt < 16; ++jt) {
            v8f acc = {};
            const __bf16* wr = wm + (size_t)(jt * 16 + ln) * Cc;
#pragma unroll
            for (int s = 0; s < 8; ++s) {
                FragB16 bf;
#pragma unroll
                for (int v = 0; v < 8; ++v)
                    bf.u[v] = *(const unsigned int*)&wr[s * 32 + a_k(v, g)];
                acc = __builtin_amdgcn_wmma_f32_16x16x32_bf16(false, af[s].v, false, bf.v,
                                                              (short)0, acc, false, false);
            }
            float bb = bias[mat * Cc + jt * 16 + ln];
#pragma unroll
            for (int r = 0; r < 8; ++r)
                om[(size_t)(r + 8 * g) * Cc + jt * 16 + ln] = (__bf16)(acc[r] + bb);
        }
    }
}

// ---------------------------------------------------------------------------
// Kernel 3: fused flash attention (Q pre-negated), async double-buffered K/V.
// Block = 128 threads (4 waves) = 64 query rows; loop over 32-key blocks of N.
// ---------------------------------------------------------------------------
__device__ __forceinline__ void stage_tile(const __bf16* Kb, const __bf16* Vb, int kt,
                                           char* ks_buf, char* vraw_buf, int t) {
    const char* kg = (const char*)(Kb + (size_t)kt * 32 * Cc);
    const char* vg = (const char*)(Vb + (size_t)kt * 32 * Cc);
#pragma unroll
    for (int j = 0; j < 8; ++j) {
        int i = j * 128 + t;
        int row = i >> 5, col = i & 31;
        async_copy16(kg + (size_t)i * 16, ks_buf + row * (KS_STRIDE * 2) + col * 16);
        async_copy16(vg + (size_t)i * 16, vraw_buf + (size_t)i * 16);
    }
}

__global__ __launch_bounds__(128) void SSAML_attn_kernel(
        const unsigned short* __restrict__ q_, const unsigned short* __restrict__ k_,
        const unsigned short* __restrict__ v_, const float* __restrict__ x,
        const float* __restrict__ gamma, float* __restrict__ out) {
    const __bf16* Q = reinterpret_cast<const __bf16*>(q_);
    const __bf16* K = reinterpret_cast<const __bf16*>(k_);
    const __bf16* V = reinterpret_cast<const __bf16*>(v_);

    extern __shared__ __align__(16) char smem[];
    __bf16* Vt = (__bf16*)(smem + OFF_VT);
    __bf16* Ps = (__bf16*)(smem + OFF_PS);
    float*  Of = (float*)smem;                    // epilogue alias

    const int b  = blockIdx.x / (Nn / 64);
    const int n0 = (blockIdx.x % (Nn / 64)) * 64;
    const int t = threadIdx.x;
    const int lane = t & 31, wave = t >> 5;
    const int g = lane >> 4, ln = lane & 15;
    const int qrow = n0 + wave * 16;
    __bf16* pw = Ps + wave * (16 * 32);

    // Q fragments held in registers for the whole kernel (8 K-steps)
    FragB16 qf[8];
    const __bf16* Qb = Q + ((size_t)b * Nn + qrow) * Cc;
#pragma unroll
    for (int s = 0; s < 8; ++s)
#pragma unroll
        for (int v = 0; v < 8; ++v)
            qf[s].u[v] = *(const unsigned int*)&Qb[(size_t)ln * Cc + s * 32 + a_k(v, g)];

    v8f O[16] = {};
    float m_r[8], l_r[8];
#pragma unroll
    for (int r = 0; r < 8; ++r) { m_r[r] = -INFINITY; l_r[r] = 0.f; }

    const __bf16* Kb = K + (size_t)b * Nn * Cc;
    const __bf16* Vb = V + (size_t)b * Nn * Cc;
    const int NT = Nn / 32;                       // 72 key tiles

    // prologue: tile 0 in flight
    stage_tile(Kb, Vb, 0, smem + OFF_KS0, smem + OFF_VR0, t);

    for (int kt = 0; kt < NT; ++kt) {
        const int cur = kt & 1;
        char* ks_cur = smem + OFF_KS0 + cur * KSB_BYTES;
        char* vr_cur = smem + OFF_VR0 + cur * VRAW_BYTES;
        __syncthreads();  // all waves done reading the other buffer from previous iteration
        if (kt + 1 < NT) {
            char* ks_nxt = smem + OFF_KS0 + (cur ^ 1) * KSB_BYTES;
            char* vr_nxt = smem + OFF_VR0 + (cur ^ 1) * VRAW_BYTES;
            stage_tile(Kb, Vb, kt + 1, ks_nxt, vr_nxt, t);
            WAIT_ASYNC(16);   // in-order: tile kt's 16 ops per wave are complete
        } else {
            WAIT_ASYNC(0);
        }
        __syncthreads();  // all waves' quarters of tile kt are in LDS

        // Transpose Vraw[cur] (key-major) -> Vt (channel-major) for PV B-fragments
        {
            const char* vr = vr_cur;
#pragma unroll
            for (int j = 0; j < 8; ++j) {
                int i = j * 128 + t;
                int row = i >> 5, col8 = (i & 31) * 8;
                union { uint4 q; __bf16 h[8]; } u;
                u.q = *(const uint4*)(vr + (size_t)i * 16);
#pragma unroll
                for (int e = 0; e < 8; ++e)
                    Vt[(col8 + e) * VT_STRIDE + row] = u.h[e];
            }
        }
        __syncthreads();  // Vt ready

        const __bf16* KsC = (const __bf16*)ks_cur;

        // S = Qneg . K^T  -> two 16x16 f32 tiles (32 keys)
        v8f S0 = {}, S1 = {};
#pragma unroll
        for (int s = 0; s < 8; ++s) {
            FragB16 b0, b1;
#pragma unroll
            for (int v = 0; v < 8; ++v) {
                b0.u[v] = *(const unsigned int*)&KsC[ln * KS_STRIDE + s * 32 + a_k(v, g)];
                b1.u[v] = *(const unsigned int*)&KsC[(16 + ln) * KS_STRIDE + s * 32 + a_k(v, g)];
            }
            S0 = __builtin_amdgcn_wmma_f32_16x16x32_bf16(false, qf[s].v, false, b0.v,
                                                         (short)0, S0, false, false);
            S1 = __builtin_amdgcn_wmma_f32_16x16x32_bf16(false, qf[s].v, false, b1.v,
                                                         (short)0, S1, false, false);
        }

        // Online softmax over this 32-key block (rows live across 16-lane groups)
        float p0[8], p1[8];
#pragma unroll
        for (int r = 0; r < 8; ++r) {
            float mx = fmaxf(S0[r], S1[r]);
#pragma unroll
            for (int o = 1; o < 16; o <<= 1) mx = fmaxf(mx, __shfl_xor(mx, o, 32));
            float mnew = fmaxf(m_r[r], mx);
            float sc = __expf(m_r[r] - mnew);
            p0[r] = __expf(S0[r] - mnew);
            p1[r] = __expf(S1[r] - mnew);
            float rs = p0[r] + p1[r];
#pragma unroll
            for (int o = 1; o < 16; o <<= 1) rs += __shfl_xor(rs, o, 32);
            l_r[r] = l_r[r] * sc + rs;
            m_r[r] = mnew;
#pragma unroll
            for (int ct = 0; ct < 16; ++ct) O[ct][r] *= sc;
        }

        // Bounce P (D-layout) through wave-private LDS into A-layout
#pragma unroll
        for (int r = 0; r < 8; ++r) {
            pw[(r + 8 * g) * 32 + ln]      = (__bf16)p0[r];
            pw[(r + 8 * g) * 32 + 16 + ln] = (__bf16)p1[r];
        }
        FragB16 pa;
#pragma unroll
        for (int v = 0; v < 8; ++v)
            pa.u[v] = *(const unsigned int*)&pw[ln * 32 + a_k(v, g)];

        // O += P . V   (16 column tiles of 16 channels)
#pragma unroll
        for (int ct = 0; ct < 16; ++ct) {
            FragB16 vf;
#pragma unroll
            for (int v = 0; v < 8; ++v)
                vf.u[v] = *(const unsigned int*)&Vt[(ct * 16 + ln) * VT_STRIDE + a_k(v, g)];
            O[ct] = __builtin_amdgcn_wmma_f32_16x16x32_bf16(false, pa.v, false, vf.v,
                                                            (short)0, O[ct], false, false);
        }
    }

    // Epilogue: bounce O/l through LDS (token-major) for coalesced global I/O:
    // out(b,c,n) = gamma * O/l + x(b,c,n)
    __syncthreads();   // done with Ks/Vraw/Vt/Ps; smem reused as Of[64][OF_STRIDE]
#pragma unroll
    for (int ct = 0; ct < 16; ++ct) {
#pragma unroll
        for (int r = 0; r < 8; ++r) {
            int tok = wave * 16 + r + 8 * g;
            Of[tok * OF_STRIDE + ct * 16 + ln] = O[ct][r] / l_r[r];
        }
    }
    __syncthreads();
    const float gm = gamma[0];
    const float* xb = x + (size_t)b * Cc * Nn;
    float* ob = out + (size_t)b * Cc * Nn;
    for (int j = 0; j < 128; ++j) {
        int idx = j * 128 + t;
        int c = idx >> 6, n = idx & 63;
        size_t gi = (size_t)c * Nn + n0 + n;
        ob[gi] = gm * Of[n * OF_STRIDE + c] + xb[gi];
    }
}

extern "C" void kernel_launch(void* const* d_in, const int* in_sizes, int n_in,
                              void* d_out, int out_size, void* d_ws, size_t ws_size,
                              hipStream_t stream) {
    const float* x  = (const float*)d_in[0];
    const float* Wq = (const float*)d_in[1];
    const float* bq = (const float*)d_in[2];
    const float* Wk = (const float*)d_in[3];
    const float* bk = (const float*)d_in[4];
    const float* Wv = (const float*)d_in[5];
    const float* bv = (const float*)d_in[6];
    const float* gm = (const float*)d_in[7];
    float* out = (float*)d_out;

    char* ws = (char*)d_ws;
    unsigned short* qkv  = (unsigned short*)ws;
    unsigned short* wbf  = (unsigned short*)(ws + 3 * QELEMS * 2);
    float*          bias = (float*)(ws + 3 * QELEMS * 2 + 3 * (size_t)Cc * Cc * 2);

    SSAML_prep_kernel<<<(3 * Cc * Cc + 255) / 256, 256, 0, stream>>>(Wq, bq, Wk, bk, Wv, bv,
                                                                     wbf, bias);
    SSAML_qkv_kernel<<<Bn * (Nn / 64), 128, 0, stream>>>(x, wbf, bias, qkv);
    SSAML_attn_kernel<<<Bn * (Nn / 64), 128, ATTN_LDS_TOTAL, stream>>>(
        qkv, qkv + QELEMS, qkv + 2 * QELEMS, x, gm, out);
}
